// MultiHeadAttention_68272800137483
// MI455X (gfx1250) — matrix operand order repaired and checked by reading, not compiled, not run on hardware
//
#include <hip/hip_runtime.h>
#include <hip/hip_bf16.h>

typedef __bf16 bf16;
typedef __attribute__((ext_vector_type(16))) __bf16 bf16x16;
typedef __attribute__((ext_vector_type(8)))  float  f32x8;

#define N_ 8
#define L_ 4096
#define S_ 4096
#define E_ 256
#define H_ 8
#define D_ 32
#define MROWS (N_ * L_)   // 32768 flattened rows

union FragAB {
    bf16x16 v;
    bf16    e[16];
    uint4   q[2];
};

__device__ __forceinline__ f32x8 wmma_bf16(bf16x16 a, bf16x16 b, f32x8 c) {
    // D = A(16x32 bf16) * B(32x16 bf16) + C(16x16 f32)
    return __builtin_amdgcn_wmma_f32_16x16x32_bf16(false, a, false, b, (short)0, c,
                                                   false, false);
}

// ---------------------------------------------------------------------------
// Weight conversion f32 -> bf16 (tiny, once per launch)
// ---------------------------------------------------------------------------
__global__ void cvt_f32_bf16(const float* __restrict__ src, bf16* __restrict__ dst, int n) {
    int i = blockIdx.x * blockDim.x + threadIdx.x;
    if (i < n) dst[i] = (bf16)src[i];
}

// ---------------------------------------------------------------------------
// Projection GEMM: Y = act(X @ W^T + b) * mask * scale   (Y bf16)
// Block: 256 threads = 8 waves; block tile = 16 rows x 256 cols;
// wave tile = 16x32 (two accumulators, A fragment reused across both WMMAs).
// ---------------------------------------------------------------------------
__global__ void proj_gemm(const float* __restrict__ X, const bf16* __restrict__ Wb,
                          const float* __restrict__ bias, const unsigned char* __restrict__ mask,
                          float scale, int doElu, bf16* __restrict__ Y)
{
    const int wave = threadIdx.x >> 5;
    const int lane = threadIdx.x & 31;
    const int hf   = lane >> 4;     // half of wave (K-split per ISA layouts)
    const int mn   = lane & 15;     // M index (A) / N index (B, C/D)
    const int rowbase = blockIdx.x * 16;
    const int colbase = wave * 32;

    // pull weight rows for this wave's columns toward the caches early
    __builtin_prefetch(Wb + (size_t)(colbase + mn) * E_, 0, 2);

    f32x8 acc0 = {}, acc1 = {};
    #pragma unroll
    for (int kb = 0; kb < E_; kb += 32) {
        // A fragment: X rows, f32 -> bf16. e<8: K=hf*8+e ; e>=8: K=16+hf*8+(e-8)
        const float* ax = X + (size_t)(rowbase + mn) * E_ + kb + hf * 8;
        float4 a0 = *(const float4*)(ax);
        float4 a1 = *(const float4*)(ax + 4);
        float4 a2 = *(const float4*)(ax + 16);
        float4 a3 = *(const float4*)(ax + 20);
        FragAB a;
        a.e[0]=(bf16)a0.x; a.e[1]=(bf16)a0.y; a.e[2]=(bf16)a0.z; a.e[3]=(bf16)a0.w;
        a.e[4]=(bf16)a1.x; a.e[5]=(bf16)a1.y; a.e[6]=(bf16)a1.z; a.e[7]=(bf16)a1.w;
        a.e[8]=(bf16)a2.x; a.e[9]=(bf16)a2.y; a.e[10]=(bf16)a2.z; a.e[11]=(bf16)a2.w;
        a.e[12]=(bf16)a3.x; a.e[13]=(bf16)a3.y; a.e[14]=(bf16)a3.z; a.e[15]=(bf16)a3.w;
        // B fragments: B[k][n] = W[col n][k]; e<8: K=hf*8+e ; e>=8: K=16+hf*8+(e-8)
        const bf16* bx0 = Wb + (size_t)(colbase + mn) * E_ + kb + hf * 8;
        const bf16* bx1 = bx0 + (size_t)16 * E_;
        FragAB b0, b1;
        b0.q[0] = *(const uint4*)(bx0);
        b0.q[1] = *(const uint4*)(bx0 + 16);
        b1.q[0] = *(const uint4*)(bx1);
        b1.q[1] = *(const uint4*)(bx1 + 16);
        acc0 = wmma_bf16(a.v, b0.v, acc0);
        acc1 = wmma_bf16(a.v, b1.v, acc1);
    }

    const int col0 = colbase + mn;
    const int col1 = col0 + 16;
    const float bc0 = bias[col0];
    const float bc1 = bias[col1];
    #pragma unroll
    for (int r = 0; r < 8; ++r) {
        int row = rowbase + r + 8 * hf;            // C/D layout: M = r + 8*half
        float m  = mask[row] ? scale : 0.0f;
        float v0 = acc0[r] + bc0;
        float v1 = acc1[r] + bc1;
        if (doElu) {
            v0 = (v0 > 0.0f) ? (v0 + 1.0f) : __expf(v0);  // elu(x)+1
            v1 = (v1 > 0.0f) ? (v1 + 1.0f) : __expf(v1);
        }
        Y[(size_t)row * E_ + col0] = (bf16)(v0 * m);
        Y[(size_t)row * E_ + col1] = (bf16)(v1 * m);
    }
}

// ---------------------------------------------------------------------------
// KV aggregation per (n,h): KV[d][v] = sum_s Kf[s,d]*Vm[s,v]; also Ksum[d].
// Stores KV transposed (KVt[v][d], bf16) so attn B-fragments are contiguous.
// One block per (n,h); 8 waves each own interleaved 32-row S-chunks.
// ---------------------------------------------------------------------------
__global__ void kv_agg(const bf16* __restrict__ Kf, const bf16* __restrict__ Vm,
                       bf16* __restrict__ KVt, float* __restrict__ Ksum)
{
    __shared__ __align__(16) unsigned char smem[33792];  // 2x [8][32][33] ushort; reused f32
    unsigned short* kb_s = (unsigned short*)smem;
    unsigned short* vb_s = (unsigned short*)(smem + 16896);

    const int nh   = blockIdx.x;           // n*H + h
    const int wave = threadIdx.x >> 5;
    const int lane = threadIdx.x & 31;
    const int hf   = lane >> 4;
    const int mn   = lane & 15;
    const size_t base = (size_t)(nh / H_) * S_ * E_ + (size_t)(nh % H_) * D_;

    unsigned short* myk = kb_s + wave * 1056;   // 32*33
    unsigned short* myv = vb_s + wave * 1056;

    f32x8 acc00 = {}, acc01 = {}, acc10 = {}, acc11 = {};

    for (int i = 0; i < 16; ++i) {
        const int sbase = (wave + 8 * i) * 32;
        // stage 32(s) x 32(d/v) chunk of Kf and Vm into this wave's LDS slice
        #pragma unroll 4
        for (int r = 0; r < 32; ++r) {
            size_t g = base + (size_t)(sbase + r) * E_ + lane;
            myk[r * 33 + lane] = ((const unsigned short*)Kf)[g];
            myv[r * 33 + lane] = ((const unsigned short*)Vm)[g];
        }
        // A = Kf^T (M=d, K=s): gather transposed; B = Vm (K=s, N=v)
        FragAB a0, a1, b0, b1;
        #pragma unroll
        for (int e = 0; e < 16; ++e) {
            int Ka = (e < 8) ? (hf * 8 + e) : (16 + hf * 8 + (e - 8));
            int Kb = Ka;
            a0.e[e] = __builtin_bit_cast(bf16, myk[Ka * 33 + mn]);
            a1.e[e] = __builtin_bit_cast(bf16, myk[Ka * 33 + 16 + mn]);
            b0.e[e] = __builtin_bit_cast(bf16, myv[Kb * 33 + mn]);
            b1.e[e] = __builtin_bit_cast(bf16, myv[Kb * 33 + 16 + mn]);
        }
        acc00 = wmma_bf16(a0.v, b0.v, acc00);
        acc01 = wmma_bf16(a0.v, b1.v, acc01);
        acc10 = wmma_bf16(a1.v, b0.v, acc10);
        acc11 = wmma_bf16(a1.v, b1.v, acc11);
    }

    // cross-wave reduction of the 32x32 tile
    __syncthreads();
    float* red = (float*)smem;                  // [8 waves][4 tiles * 256]
    {
        float* my = red + wave * 1024;
        #pragma unroll
        for (int r = 0; r < 8; ++r) {
            my[0 * 256 + r * 32 + lane] = acc00[r];
            my[1 * 256 + r * 32 + lane] = acc01[r];
            my[2 * 256 + r * 32 + lane] = acc10[r];
            my[3 * 256 + r * 32 + lane] = acc11[r];
        }
    }
    __syncthreads();
    const int t = threadIdx.x;
    #pragma unroll
    for (int j = 0; j < 4; ++j) {
        int o = j * 256 + t;
        float s = 0.0f;
        #pragma unroll
        for (int w2 = 0; w2 < 8; ++w2) s += red[w2 * 1024 + o];
        int t4 = o >> 8, rr = (o >> 5) & 7, ln = o & 31;
        int dt = t4 >> 1, vt = t4 & 1;
        int d = dt * 16 + rr + 8 * (ln >> 4);
        int vv = vt * 16 + (ln & 15);
        KVt[(size_t)nh * 1024 + vv * 32 + d] = (bf16)s;   // transposed store
    }

    // Ksum[d] = sum_s Kf[s,d]
    __syncthreads();
    float* kpart = (float*)smem;
    {
        int d = t & 31, grp = t >> 5;
        float p = 0.0f;
        const bf16* kcol = Kf + base + d;
        for (int s2 = grp * 512; s2 < grp * 512 + 512; ++s2)
            p += (float)kcol[(size_t)s2 * E_];
        kpart[t] = p;
    }
    __syncthreads();
    if (t < 32) {
        float s = 0.0f;
        #pragma unroll
        for (int g = 0; g < 8; ++g) s += kpart[g * 32 + t];
        Ksum[nh * 32 + t] = s;
    }
}

// ---------------------------------------------------------------------------
// out = (Qf @ KV) * Z * S  per head; Z = 1/(Qf . Ksum + eps).
// Block = 16 L-rows, wave index = head. Single K=32 WMMA per 16x16 tile.
// ---------------------------------------------------------------------------
__global__ void attn_apply(const bf16* __restrict__ Qf, const bf16* __restrict__ KVt,
                           const float* __restrict__ Ksum, bf16* __restrict__ OutPre)
{
    __shared__ float zbuf[128];
    const int wave = threadIdx.x >> 5;          // head
    const int lane = threadIdx.x & 31;
    const int hf   = lane >> 4;
    const int mn   = lane & 15;
    const int lbase = blockIdx.x * 16;          // flattened n*L + l row base
    const int n  = lbase / L_;
    const int nh = n * H_ + wave;

    // A = Qf tile (16 rows x 32 d), contiguous halves
    FragAB a;
    const bf16* ax = Qf + (size_t)(lbase + mn) * E_ + wave * D_ + hf * 16;
    a.q[0] = *(const uint4*)(ax);
    a.q[1] = *(const uint4*)(ax + 8);
    // B = KV (32 d x 16 v) from transposed store: row v, K contiguous
    const bf16* bx = KVt + (size_t)nh * 1024 + (size_t)mn * 32 + hf * 16;
    FragAB b0, b1;
    b0.q[0] = *(const uint4*)(bx);
    b0.q[1] = *(const uint4*)(bx + 8);
    b1.q[0] = *(const uint4*)(bx + 512);
    b1.q[1] = *(const uint4*)(bx + 512 + 8);

    f32x8 c0 = {}, c1 = {};
    c0 = wmma_bf16(a.v, b0.v, c0);
    c1 = wmma_bf16(a.v, b1.v, c1);

    // normalizer per row
    if (lane < 16) {
        const bf16*  qrow = Qf + (size_t)(lbase + lane) * E_ + wave * D_;
        const float* ks   = Ksum + nh * 32;
        float dot = 0.0f;
        #pragma unroll
        for (int d = 0; d < 32; ++d) dot += (float)qrow[d] * ks[d];
        zbuf[wave * 16 + lane] = (float)S_ / (dot + 1e-6f);
    }
    __syncthreads();

    #pragma unroll
    for (int r = 0; r < 8; ++r) {
        int m = r + 8 * hf;
        float z = zbuf[wave * 16 + m];
        size_t o = (size_t)(lbase + m) * E_ + wave * D_ + mn;
        OutPre[o]      = (bf16)(c0[r] * z);
        OutPre[o + 16] = (bf16)(c1[r] * z);
    }
}

// ---------------------------------------------------------------------------
// Merge GEMM: d_out = OutPre(bf16) @ Wm^T  (f32 out, no bias/act)
// Same 16x32 wave tiling as proj_gemm.
// ---------------------------------------------------------------------------
__global__ void merge_gemm(const bf16* __restrict__ X, const bf16* __restrict__ Wb,
                           float* __restrict__ Out)
{
    const int wave = threadIdx.x >> 5;
    const int lane = threadIdx.x & 31;
    const int hf   = lane >> 4;
    const int mn   = lane & 15;
    const int rowbase = blockIdx.x * 16;
    const int colbase = wave * 32;

    __builtin_prefetch(Wb + (size_t)(colbase + mn) * E_, 0, 2);

    f32x8 acc0 = {}, acc1 = {};
    #pragma unroll
    for (int kb = 0; kb < E_; kb += 32) {
        const bf16* ax = X + (size_t)(rowbase + mn) * E_ + kb + hf * 16;
        FragAB a;
        a.q[0] = *(const uint4*)(ax);        // K = hf*16 + 0..7
        a.q[1] = *(const uint4*)(ax + 8);    // K = hf*16 + 8..15
        const bf16* bx0 = Wb + (size_t)(colbase + mn) * E_ + kb + hf * 16;
        const bf16* bx1 = bx0 + (size_t)16 * E_;
        FragAB b0, b1;
        b0.q[0] = *(const uint4*)(bx0);
        b0.q[1] = *(const uint4*)(bx0 + 8);
        b1.q[0] = *(const uint4*)(bx1);
        b1.q[1] = *(const uint4*)(bx1 + 8);
        acc0 = wmma_bf16(a.v, b0.v, acc0);
        acc1 = wmma_bf16(a.v, b1.v, acc1);
    }
    const int col0 = colbase + mn;
    #pragma unroll
    for (int r = 0; r < 8; ++r) {
        int row = rowbase + r + 8 * hf;
        Out[(size_t)row * E_ + col0]      = acc0[r];
        Out[(size_t)row * E_ + col0 + 16] = acc1[r];
    }
}

// ---------------------------------------------------------------------------
extern "C" void kernel_launch(void* const* d_in, const int* in_sizes, int n_in,
                              void* d_out, int out_size, void* d_ws, size_t ws_size,
                              hipStream_t stream)
{
    const float* q  = (const float*)d_in[0];
    const float* k  = (const float*)d_in[1];
    const float* v  = (const float*)d_in[2];
    const unsigned char* qmask  = (const unsigned char*)d_in[3];  // jax bool = 1 byte
    const unsigned char* kvmask = (const unsigned char*)d_in[4];
    const float* Wq = (const float*)d_in[5];
    const float* bq = (const float*)d_in[6];
    const float* Wk = (const float*)d_in[7];
    const float* bk = (const float*)d_in[8];
    const float* Wv = (const float*)d_in[9];
    const float* bv = (const float*)d_in[10];
    const float* Wm = (const float*)d_in[11];
    float* out = (float*)d_out;

    char* ws = (char*)d_ws;
    bf16* Wq_b = (bf16*)ws;  ws += (size_t)E_ * E_ * 2;
    bf16* Wk_b = (bf16*)ws;  ws += (size_t)E_ * E_ * 2;
    bf16* Wv_b = (bf16*)ws;  ws += (size_t)E_ * E_ * 2;
    bf16* Wm_b = (bf16*)ws;  ws += (size_t)E_ * E_ * 2;
    bf16* Qf   = (bf16*)ws;  ws += (size_t)MROWS * E_ * 2;
    bf16* Kf   = (bf16*)ws;  ws += (size_t)MROWS * E_ * 2;
    bf16* Vm   = (bf16*)ws;  ws += (size_t)MROWS * E_ * 2;
    bf16* OutP = (bf16*)ws;  ws += (size_t)MROWS * E_ * 2;
    bf16* KVt  = (bf16*)ws;  ws += (size_t)N_ * H_ * D_ * D_ * 2;
    float* Ksum = (float*)ws;

    cvt_f32_bf16<<<dim3(E_ * E_ / 256), dim3(256), 0, stream>>>(Wq, Wq_b, E_ * E_);
    cvt_f32_bf16<<<dim3(E_ * E_ / 256), dim3(256), 0, stream>>>(Wk, Wk_b, E_ * E_);
    cvt_f32_bf16<<<dim3(E_ * E_ / 256), dim3(256), 0, stream>>>(Wv, Wv_b, E_ * E_);
    cvt_f32_bf16<<<dim3(E_ * E_ / 256), dim3(256), 0, stream>>>(Wm, Wm_b, E_ * E_);

    dim3 gProj(MROWS / 16);
    proj_gemm<<<gProj, 256, 0, stream>>>(q, Wq_b, bq, qmask,  1.0f,             1, Qf);
    proj_gemm<<<gProj, 256, 0, stream>>>(k, Wk_b, bk, kvmask, 1.0f,             1, Kf);
    proj_gemm<<<gProj, 256, 0, stream>>>(v, Wv_b, bv, kvmask, 1.0f / (float)S_, 0, Vm);

    kv_agg<<<dim3(N_ * H_), dim3(256), 0, stream>>>(Kf, Vm, KVt, Ksum);
    attn_apply<<<dim3(MROWS / 16), 256, 0, stream>>>(Qf, KVt, Ksum, OutP);
    merge_gemm<<<gProj, 256, 0, stream>>>(OutP, Wm_b, out);
}
